// TimeMixing_7215545057483
// MI455X (gfx1250) — matrix operand order repaired and checked by reading, not compiled, not run on hardware
//
#include <hip/hip_runtime.h>
#include <hip/hip_bf16.h>

// Problem constants (from the reference):
#define LSEQ   2048
#define BB     8
#define DD     1024
#define MM     (LSEQ * BB)        // 16384 rows for all GEMMs
#define NCH    (BB * DD)          // 8192 scan channels
#define NCHUNK 16
#define CLEN   (LSEQ / NCHUNK)    // 128 steps per chunk

typedef __attribute__((ext_vector_type(16))) _Float16 v16h;
typedef __attribute__((ext_vector_type(8)))  _Float16 v8h;
typedef __attribute__((ext_vector_type(8)))  float    v8f;

union FragU { v16h v; v8h h[2]; };

// ---------------------------------------------------------------------------
// f32 -> f16 convert (weights)
// ---------------------------------------------------------------------------
__global__ __launch_bounds__(256)
void f32_to_f16_kernel(const float* __restrict__ in, _Float16* __restrict__ out, int n) {
    int i = blockIdx.x * blockDim.x + threadIdx.x;
    if (i < n) out[i] = (_Float16)in[i];
}

// ---------------------------------------------------------------------------
// Fused token-shift mix: reads x once, writes all three f16 mixes.
//   xk = mu_k*x_t + (1-mu_k)*x_{t-1}   (and same for v, r; zero initial prev)
// ---------------------------------------------------------------------------
__global__ __launch_bounds__(256)
void mix3_kernel(const float* __restrict__ x,
                 const float* __restrict__ mu_k, const float* __restrict__ mu_v,
                 const float* __restrict__ mu_r,
                 _Float16* __restrict__ xk, _Float16* __restrict__ xv,
                 _Float16* __restrict__ xr, int total) {
    int idx = blockIdx.x * blockDim.x + threadIdx.x;
    if (idx >= total) return;
    int d   = idx & (DD - 1);
    int row = idx >> 10;                 // row = l*B + b
    float xc = x[idx];
    float xp = (row >= BB) ? x[idx - BB * DD] : 0.0f;
    float mk = mu_k[d], mv = mu_v[d], mr = mu_r[d];
    xk[idx] = (_Float16)(mk * xc + (1.0f - mk) * xp);
    xv[idx] = (_Float16)(mv * xc + (1.0f - mv) * xp);
    xr[idx] = (_Float16)(mr * xc + (1.0f - mr) * xp);
}

// ---------------------------------------------------------------------------
// WMMA GEMM: C[M,N] = A[M,K] * W[N,K]^T, A/W f16 row-major, C = OutT.
// Block = 128 threads = 4 waves (2x2), each wave owns a 64x64 tile
// (4x4 grid of 16x16 WMMA tiles), K-loop in steps of 32, f32 accumulate.
// ---------------------------------------------------------------------------
template <typename OutT>
__global__ __launch_bounds__(128)
void wmma_gemm_kernel(const _Float16* __restrict__ A,
                      const _Float16* __restrict__ W,
                      OutT* __restrict__ C,
                      int M, int N, int K) {
    const int lane  = threadIdx.x & 31;
    const int wave  = threadIdx.x >> 5;      // 0..3
    const int waveM = wave >> 1;
    const int waveN = wave & 1;
    const int m0 = blockIdx.y * 128 + waveM * 64;
    const int n0 = blockIdx.x * 128 + waveN * 64;

    const int half = lane >> 4;              // 0 or 1
    const int r16  = lane & 15;

    v8f acc[4][4];
#pragma unroll
    for (int i = 0; i < 4; ++i)
#pragma unroll
        for (int j = 0; j < 4; ++j)
#pragma unroll
            for (int e = 0; e < 8; ++e) acc[i][j][e] = 0.0f;

    const _Float16* Abase = A + (size_t)(m0 + r16) * K;
    const _Float16* Wbase = W + (size_t)(n0 + r16) * K;

    for (int k0 = 0; k0 < K; k0 += 32) {
        __builtin_prefetch((const void*)(Abase + k0 + 64), 0, 3);
        __builtin_prefetch((const void*)(Wbase + k0 + 64), 0, 3);

        FragU afrag[4], bfrag[4];
#pragma unroll
        for (int i = 0; i < 4; ++i) {
            // A fragment (16x32): lane m: K = k0+half*8+[0..7], k0+16+half*8+[0..7]
            const _Float16* pa = Abase + (size_t)(i * 16) * K + k0 + half * 8;
            afrag[i].h[0] = *(const v8h*)(pa);
            afrag[i].h[1] = *(const v8h*)(pa + 16);
            // B fragment (32x16): lane n: K = k0+half*8+[0..7], k0+16+half*8+[0..7] of W row n
            const _Float16* pb = Wbase + (size_t)(i * 16) * K + k0 + half * 8;
            bfrag[i].h[0] = *(const v8h*)(pb);
            bfrag[i].h[1] = *(const v8h*)(pb + 16);
        }
#pragma unroll
        for (int i = 0; i < 4; ++i)
#pragma unroll
            for (int j = 0; j < 4; ++j)
                acc[i][j] = __builtin_amdgcn_wmma_f32_16x16x32_f16(
                    false, afrag[i].v, false, bfrag[j].v,
                    (short)0, acc[i][j], false, false);
    }

    // C/D layout: element e of lane L -> row = (L>>4)*8 + e, col = L&15.
#pragma unroll
    for (int i = 0; i < 4; ++i)
#pragma unroll
        for (int j = 0; j < 4; ++j) {
            const int col = n0 + j * 16 + r16;
#pragma unroll
            for (int e = 0; e < 8; ++e) {
                const int row = m0 + i * 16 + half * 8 + e;
                C[(size_t)row * N + col] = (OutT)acc[i][j][e];
            }
        }
}

// ---------------------------------------------------------------------------
// Chunked WKV scan (linear recurrence, split for parallelism):
//   A_t = exp_w * A_{t-1} + exp(k_t);   den_t = A_{t-1} + e^u * e^{k_t}
// Stage A: per (channel, chunk) local carries with zero initial state.
// Stage B: per channel, propagate carries across the 16 chunks.
// Stage C: per (channel, chunk) replay from corrected state, write h.
// ---------------------------------------------------------------------------
__global__ __launch_bounds__(256)
void wkv_carry_kernel(const _Float16* __restrict__ kbuf,
                      const _Float16* __restrict__ vbuf,
                      const float* __restrict__ w,
                      float* __restrict__ carry_den, float* __restrict__ carry_num) {
    int tid = blockIdx.x * blockDim.x + threadIdx.x;   // 0 .. NCH*NCHUNK-1
    if (tid >= NCH * NCHUNK) return;
    int c     = tid & (NCH - 1);
    int chunk = tid >> 13;                 // / NCH
    int d     = c & (DD - 1);
    const float exp_w = __expf(-__expf(w[d]));
    float aden = 0.0f, anum = 0.0f;
    int t0 = chunk * CLEN;
    for (int t = t0; t < t0 + CLEN; ++t) {
        size_t idx = (size_t)t * NCH + c;
        float ek  = __expf((float)kbuf[idx]);
        float ekv = ek * (float)vbuf[idx];
        aden = aden * exp_w + ek;
        anum = anum * exp_w + ekv;
    }
    carry_den[tid] = aden;
    carry_num[tid] = anum;
}

__global__ __launch_bounds__(256)
void wkv_propagate_kernel(const float* __restrict__ w,
                          float* __restrict__ carry_den, float* __restrict__ carry_num) {
    int c = blockIdx.x * blockDim.x + threadIdx.x;     // 0 .. NCH-1
    if (c >= NCH) return;
    int d = c & (DD - 1);
    const float w_neg  = -__expf(w[d]);
    const float decayC = __expf(w_neg * (float)CLEN);  // exp_w^CLEN
    float aden = 0.0f, anum = 0.0f;                    // A_{-1} entering chunk 0
    for (int j = 0; j < NCHUNK; ++j) {
        int slot = j * NCH + c;
        float cd = carry_den[slot];
        float cn = carry_num[slot];
        carry_den[slot] = aden;                        // initial state for chunk j
        carry_num[slot] = anum;
        aden = decayC * aden + cd;
        anum = decayC * anum + cn;
    }
}

__global__ __launch_bounds__(256)
void wkv_output_kernel(const _Float16* __restrict__ kbuf,
                       const _Float16* __restrict__ vbuf,
                       const _Float16* __restrict__ rbuf,
                       const float* __restrict__ w, const float* __restrict__ u,
                       const float* __restrict__ carry_den,
                       const float* __restrict__ carry_num,
                       _Float16* __restrict__ h) {
    int tid = blockIdx.x * blockDim.x + threadIdx.x;
    if (tid >= NCH * NCHUNK) return;
    int c     = tid & (NCH - 1);
    int chunk = tid >> 13;
    int d     = c & (DD - 1);
    const float exp_w = __expf(-__expf(w[d]));
    const float exp_u = __expf(u[d]);
    float aden = carry_den[tid];     // A_{t0-1}
    float anum = carry_num[tid];
    int t0 = chunk * CLEN;
    for (int t = t0; t < t0 + CLEN; ++t) {
        size_t idx = (size_t)t * NCH + c;
        float ek  = __expf((float)kbuf[idx]);
        float ekv = ek * (float)vbuf[idx];
        float den = aden + ek * exp_u;
        float num = anum + ekv * exp_u;
        float sr  = 1.0f / (1.0f + __expf(-(float)rbuf[idx]));
        h[idx] = (_Float16)(sr * num / den);
        aden = aden * exp_w + ek;
        anum = anum * exp_w + ekv;
    }
}

// ---------------------------------------------------------------------------
// Launch
// ---------------------------------------------------------------------------
extern "C" void kernel_launch(void* const* d_in, const int* in_sizes, int n_in,
                              void* d_out, int out_size, void* d_ws, size_t ws_size,
                              hipStream_t stream) {
    const float* x     = (const float*)d_in[0];
    const float* mu_k  = (const float*)d_in[1];
    const float* mu_v  = (const float*)d_in[2];
    const float* mu_r  = (const float*)d_in[3];
    const float* Wk    = (const float*)d_in[4];
    const float* Wv    = (const float*)d_in[5];
    const float* Wr    = (const float*)d_in[6];
    const float* Wout  = (const float*)d_in[7];
    const float* w     = (const float*)d_in[8];
    const float* u     = (const float*)d_in[9];
    float* out = (float*)d_out;

    // Workspace layout (~233 MiB total):
    char* p = (char*)d_ws;
    _Float16* Wk16   = (_Float16*)p;  p += (size_t)DD * DD * 2;   // 2 MiB each
    _Float16* Wv16   = (_Float16*)p;  p += (size_t)DD * DD * 2;
    _Float16* Wr16   = (_Float16*)p;  p += (size_t)DD * DD * 2;
    _Float16* Wout16 = (_Float16*)p;  p += (size_t)DD * DD * 2;
    _Float16* xkbuf  = (_Float16*)p;  p += (size_t)MM * DD * 2;   // 32 MiB each
    _Float16* xvbuf  = (_Float16*)p;  p += (size_t)MM * DD * 2;
    _Float16* xrbuf  = (_Float16*)p;  p += (size_t)MM * DD * 2;
    _Float16* kbuf   = (_Float16*)p;  p += (size_t)MM * DD * 2;
    _Float16* vbuf   = (_Float16*)p;  p += (size_t)MM * DD * 2;
    _Float16* rbuf   = (_Float16*)p;  p += (size_t)MM * DD * 2;
    _Float16* hbuf   = (_Float16*)p;  p += (size_t)MM * DD * 2;
    float* carry_den = (float*)p;     p += (size_t)NCH * NCHUNK * 4;  // 0.5 MiB each
    float* carry_num = (float*)p;     p += (size_t)NCH * NCHUNK * 4;
    (void)ws_size; (void)in_sizes; (void)n_in; (void)out_size;

    const int wn = DD * DD;
    f32_to_f16_kernel<<<(wn + 255) / 256, 256, 0, stream>>>(Wk,   Wk16,   wn);
    f32_to_f16_kernel<<<(wn + 255) / 256, 256, 0, stream>>>(Wv,   Wv16,   wn);
    f32_to_f16_kernel<<<(wn + 255) / 256, 256, 0, stream>>>(Wr,   Wr16,   wn);
    f32_to_f16_kernel<<<(wn + 255) / 256, 256, 0, stream>>>(Wout, Wout16, wn);

    const int total = MM * DD;
    dim3 gemmGrid(DD / 128, MM / 128);   // (8, 128)
    dim3 gemmBlock(128);

    // Fused token-shift (x read once, three f16 outputs)
    mix3_kernel<<<(total + 255) / 256, 256, 0, stream>>>(
        x, mu_k, mu_v, mu_r, xkbuf, xvbuf, xrbuf, total);

    // Projections (f32 accumulate, f16 stored)
    wmma_gemm_kernel<_Float16><<<gemmGrid, gemmBlock, 0, stream>>>(xkbuf, Wk16, kbuf, MM, DD, DD);
    wmma_gemm_kernel<_Float16><<<gemmGrid, gemmBlock, 0, stream>>>(xvbuf, Wv16, vbuf, MM, DD, DD);
    wmma_gemm_kernel<_Float16><<<gemmGrid, gemmBlock, 0, stream>>>(xrbuf, Wr16, rbuf, MM, DD, DD);

    // Chunk-parallel WKV scan + gating -> h (f16)
    const int nA = NCH * NCHUNK;
    wkv_carry_kernel<<<(nA + 255) / 256, 256, 0, stream>>>(kbuf, vbuf, w, carry_den, carry_num);
    wkv_propagate_kernel<<<(NCH + 255) / 256, 256, 0, stream>>>(w, carry_den, carry_num);
    wkv_output_kernel<<<(nA + 255) / 256, 256, 0, stream>>>(
        kbuf, vbuf, rbuf, w, u, carry_den, carry_num, hbuf);

    // out = h @ W_out^T  (f32 to d_out)
    wmma_gemm_kernel<float><<<gemmGrid, gemmBlock, 0, stream>>>(hbuf, Wout16, out, MM, DD, DD);
}
